// ResnetECPos_29480655520341
// MI455X (gfx1250) — compile-verified, matrix operand order checked by reading, not run on hardware
//
#include <hip/hip_runtime.h>

// ---------------------------------------------------------------------------
// Types for CDNA5 WMMA (wave32): bf16 A/B fragments (16 halves = 8 VGPRs),
// f32 C/D accumulator (8 VGPRs).
// ---------------------------------------------------------------------------
typedef __attribute__((ext_vector_type(16))) __bf16 v16bf;
typedef __attribute__((ext_vector_type(8)))  float  v8f;

#define THREADS 256
#define TM 32              // edge rows (or node rows) per workgroup
#define KP_MAX 544         // max padded K (2*259 -> 544 = 17*32)

static constexpr int NN = 16384;    // nodes
static constexpr int NE = 262144;   // edges
static constexpr int HID = 128;
static constexpr int FSTRIDE = 260; // feat row stride: 128 net + 128 pooled + 3 pos + 1 pad

// ---------------------------------------------------------------------------
// WMMA helpers
// ---------------------------------------------------------------------------
__device__ __forceinline__ v8f wmma_bf16(v16bf a, v16bf b, v8f c) {
  // (neg_a, A, neg_b, B, c_mod, C, reuse_a, reuse_b)
  return __builtin_amdgcn_wmma_f32_16x16x32_bf16(false, a, false, b, (short)0, c,
                                                 false, false);
}

// A fragment (16x32 bf16, row-major LDS tile with `stride` elements/row).
// ISA 7.12.2: lane m=l%16 holds row, lanes<16 K={0..7,16..23}, lanes>=16 +8.
__device__ __forceinline__ v16bf load_a_frag(const __bf16* base, int rowBase,
                                             int stride, int kBase, int lane) {
  const int m  = rowBase + (lane & 15);
  const int hl = (lane >> 4) & 1;
  const __bf16* rp = base + m * stride + kBase + hl * 8;
  v16bf a;
#pragma unroll
  for (int jj = 0; jj < 8; ++jj) {
    const int k = (jj < 4) ? (2 * jj) : (2 * jj + 8);  // 0,2,4,6,16,18,20,22
    a[2 * jj]     = rp[k];
    a[2 * jj + 1] = rp[k + 1];
  }
  return a;
}

// B fragment from pre-packed weights: layout [kb][n16][lane][16j] contiguous,
// so each lane issues one 32-byte load.
__device__ __forceinline__ v16bf load_b_frag(const __bf16* wp, int kb, int n16,
                                             int lane) {
  return *(const v16bf*)(wp + (size_t)((((kb << 3) + n16) << 5) + lane) * 16);
}

// Native CDNA5 float atomic max (no-return, device scope).
__device__ __forceinline__ void atomic_fmax_f32(float* addr, float v) {
  asm volatile("global_atomic_max_num_f32 %0, %1, off scope:SCOPE_DEV"
               :: "v"(addr), "v"(v)
               : "memory");
}

// ---------------------------------------------------------------------------
// Weight repack: f32 [F2 x 128] row-major -> bf16 WMMA fragment order.
// out[kb][n16][lane][j], k = kb*32 + hl*8 + j + (j>=8?8:0), n = n16*16 + lane%16
// ---------------------------------------------------------------------------
__global__ void pack_weights_kernel(const float* __restrict__ W,
                                    __bf16* __restrict__ out, int F2, int KB) {
  int tid = blockIdx.x * blockDim.x + threadIdx.x;
  int total = KB << 12;  // KB*8*32*16
  if (tid >= total) return;
  int j    = tid & 15;
  int lane = (tid >> 4) & 31;
  int n16  = (tid >> 9) & 7;
  int kb   = tid >> 12;
  int hl   = lane >> 4;
  int k = kb * 32 + hl * 8 + j + ((j >= 8) ? 8 : 0);
  int n = n16 * 16 + (lane & 15);
  float v = (k < F2) ? W[(size_t)k * HID + n] : 0.0f;
  out[tid] = (__bf16)v;
}

// ---------------------------------------------------------------------------
// net0 = p @ fc_pos_w + b   (K=3; VALU is the right tool)
// ---------------------------------------------------------------------------
__global__ void fcpos_kernel(const float* __restrict__ p,
                             const float* __restrict__ w,
                             const float* __restrict__ b,
                             float* __restrict__ out) {
  int tid = blockIdx.x * blockDim.x + threadIdx.x;  // NN*256
  int n = tid >> 8, c = tid & 255;
  float v = b[c];
  v = fmaf(p[n * 3 + 0], w[0 * 256 + c], v);
  v = fmaf(p[n * 3 + 1], w[1 * 256 + c], v);
  v = fmaf(p[n * 3 + 2], w[2 * 256 + c], v);
  out[tid] = v;
}

__global__ void init_neg_inf_kernel(float* __restrict__ acc, int n) {
  int tid = blockIdx.x * blockDim.x + threadIdx.x;
  if (tid < n) acc[tid] = -__builtin_inff();
}

// ---------------------------------------------------------------------------
// EdgeConv message GEMMs + segment-max aggregation.
// Per WG: 32 edges; LDS stages raw & relu'd cat([x_i, x_j-x_i]) in bf16.
// Wave w owns output cols [16w,16w+16) and both 16-row tiles:
//   h  = relu(A_relu @ W0 + b0)   (accumulated in K loop, tile also for short)
//   s  = A_raw @ Ws               (same K loop)
//   msg= s + (h @ W1 + b1)        (second K loop over staged H)
//   acc[dst] = max(acc[dst], msg) via global_atomic_max_num_f32
// ---------------------------------------------------------------------------
__global__ void __launch_bounds__(THREADS)
edgeconv_kernel(const float* __restrict__ feat, int fstride, int F,
                const int* __restrict__ src, const int* __restrict__ dst,
                const __bf16* __restrict__ w0p, const __bf16* __restrict__ wsp,
                const __bf16* __restrict__ w1p,
                const float* __restrict__ b0, const float* __restrict__ b1,
                int KB, float* __restrict__ acc_out) {
  __shared__ __bf16 Araw[TM * KP_MAX];
  __shared__ __bf16 Arelu[TM * KP_MAX];
  __shared__ __bf16 Htile[TM * HID];

  const int Kp = KB * 32;
  const int e0 = blockIdx.x * TM;
  const int tid = threadIdx.x;
  const int F2 = 2 * F;

  // ---- gather: coalesced per-row feature reads, uniform index loads ----
  for (int r = 0; r < TM; ++r) {
    const int e = e0 + r;
    const int di = dst[e];
    const int sj = src[e];
    const float* fi = feat + (size_t)di * fstride;
    const float* fj = feat + (size_t)sj * fstride;
    for (int k = tid; k < Kp; k += THREADS) {
      float v;
      if (k < F)        v = fi[k];
      else if (k < F2)  v = fj[k - F] - fi[k - F];
      else              v = 0.0f;
      Araw[r * Kp + k]  = (__bf16)v;
      Arelu[r * Kp + k] = (__bf16)fmaxf(v, 0.0f);
    }
  }
  __syncthreads();

  const int lane = tid & 31;
  const int n16  = tid >> 5;           // wave id == output col tile

  v8f accH0 = {}, accH1 = {}, accS0 = {}, accS1 = {};

  for (int kb = 0; kb < KB; ++kb) {
    const int kBase = kb * 32;
    v16bf aR0 = load_a_frag(Arelu, 0,  Kp, kBase, lane);
    v16bf aR1 = load_a_frag(Arelu, 16, Kp, kBase, lane);
    v16bf aW0 = load_a_frag(Araw,  0,  Kp, kBase, lane);
    v16bf aW1 = load_a_frag(Araw,  16, Kp, kBase, lane);
    v16bf bF  = load_b_frag(w0p, kb, n16, lane);
    v16bf bS  = load_b_frag(wsp, kb, n16, lane);
    accH0 = wmma_bf16(aR0, bF, accH0);
    accH1 = wmma_bf16(aR1, bF, accH1);
    accS0 = wmma_bf16(aW0, bS, accS0);
    accS1 = wmma_bf16(aW1, bS, accS1);
  }

  // bias + relu -> stage H (bf16) for fc1
  const int colBase = n16 * 16 + (lane & 15);
  const int rowOff  = (lane >> 4) * 8;
  const float bias0 = b0[colBase];
#pragma unroll
  for (int r = 0; r < 8; ++r) {
    Htile[(rowOff + r) * HID + colBase]      = (__bf16)fmaxf(accH0[r] + bias0, 0.0f);
    Htile[(16 + rowOff + r) * HID + colBase] = (__bf16)fmaxf(accH1[r] + bias0, 0.0f);
  }
  __syncthreads();

  // fc1 accumulates on top of shortcut accumulators
  for (int kb = 0; kb < HID / 32; ++kb) {
    v16bf aH0 = load_a_frag(Htile, 0,  HID, kb * 32, lane);
    v16bf aH1 = load_a_frag(Htile, 16, HID, kb * 32, lane);
    v16bf bW  = load_b_frag(w1p, kb, n16, lane);
    accS0 = wmma_bf16(aH0, bW, accS0);
    accS1 = wmma_bf16(aH1, bW, accS1);
  }

  const float bias1 = b1[colBase];
#pragma unroll
  for (int r = 0; r < 8; ++r) {
    const int er0 = e0 + rowOff + r;
    const int er1 = e0 + 16 + rowOff + r;
    atomic_fmax_f32(&acc_out[(size_t)dst[er0] * HID + colBase], accS0[r] + bias1);
    atomic_fmax_f32(&acc_out[(size_t)dst[er1] * HID + colBase], accS1[r] + bias1);
  }
}

// ---------------------------------------------------------------------------
// Finite-clamp aggregated max -> net; build feat = [net, rowmax(net), p, 0]
// ---------------------------------------------------------------------------
__global__ void __launch_bounds__(128)
fixup_kernel(const float* __restrict__ acc, const float* __restrict__ p,
             float* __restrict__ net, float* __restrict__ feat) {
  __shared__ float red[128];
  const int nid = blockIdx.x, t = threadIdx.x;
  float v = acc[(size_t)nid * HID + t];
  if (!__builtin_isfinite(v)) v = 0.0f;
  net[(size_t)nid * HID + t] = v;
  red[t] = v;
  __syncthreads();
#pragma unroll
  for (int s = 64; s > 0; s >>= 1) {
    if (t < s) red[t] = fmaxf(red[t], red[t + s]);
    __syncthreads();
  }
  const float mx = red[0];
  float* fr = feat + (size_t)nid * FSTRIDE;
  fr[t] = v;
  fr[HID + t] = mx;
  if (t < 3) fr[2 * HID + t] = p[nid * 3 + t];
  if (t == 3) fr[2 * HID + 3] = 0.0f;
}

// ---------------------------------------------------------------------------
// c = relu(net) @ fc_c + b   (WMMA, K=128)
// ---------------------------------------------------------------------------
__global__ void __launch_bounds__(THREADS)
final_wmma_kernel(const float* __restrict__ net, const __bf16* __restrict__ wcp,
                  const float* __restrict__ bc, float* __restrict__ out) {
  __shared__ __bf16 A[TM * HID];
  const int n0 = blockIdx.x * TM;
  const int tid = threadIdx.x;
  for (int idx = tid; idx < TM * HID; idx += THREADS)
    A[idx] = (__bf16)fmaxf(net[(size_t)n0 * HID + idx], 0.0f);
  __syncthreads();

  const int lane = tid & 31;
  const int n16  = tid >> 5;
  v8f c0 = {}, c1 = {};
  for (int kb = 0; kb < HID / 32; ++kb) {
    v16bf a0 = load_a_frag(A, 0,  HID, kb * 32, lane);
    v16bf a1 = load_a_frag(A, 16, HID, kb * 32, lane);
    v16bf b  = load_b_frag(wcp, kb, n16, lane);
    c0 = wmma_bf16(a0, b, c0);
    c1 = wmma_bf16(a1, b, c1);
  }
  const int col = n16 * 16 + (lane & 15);
  const int rowOff = (lane >> 4) * 8;
  const float bias = bc[col];
#pragma unroll
  for (int r = 0; r < 8; ++r) {
    out[(size_t)(n0 + rowOff + r) * HID + col]      = c0[r] + bias;
    out[(size_t)(n0 + 16 + rowOff + r) * HID + col] = c1[r] + bias;
  }
}

// ---------------------------------------------------------------------------
// Host orchestration
// ---------------------------------------------------------------------------
extern "C" void kernel_launch(void* const* d_in, const int* in_sizes, int n_in,
                              void* d_out, int out_size, void* d_ws,
                              size_t ws_size, hipStream_t stream) {
  (void)in_sizes; (void)n_in; (void)out_size; (void)ws_size;

  const float* p     = (const float*)d_in[0];
  const int*   ei    = (const int*)d_in[1];
  const float* fposw = (const float*)d_in[2];
  const float* fposb = (const float*)d_in[3];
  const float* fcw   = (const float*)d_in[4];
  const float* fcb   = (const float*)d_in[5];
  const float *w0[5], *b0[5], *w1[5], *b1[5], *wsc[5];
  for (int i = 0; i < 5; ++i) {
    w0[i]  = (const float*)d_in[6 + 5 * i + 0];
    b0[i]  = (const float*)d_in[6 + 5 * i + 1];
    w1[i]  = (const float*)d_in[6 + 5 * i + 2];
    b1[i]  = (const float*)d_in[6 + 5 * i + 3];
    wsc[i] = (const float*)d_in[6 + 5 * i + 4];
  }
  const int* src = ei;
  const int* dst = ei + NE;

  // ---- workspace carve ----
  char* wsb = (char*)d_ws;
  size_t off = 0;
  auto carve = [&](size_t bytes) -> void* {
    void* r = (void*)(wsb + off);
    off = (off + bytes + 255) & ~(size_t)255;
    return r;
  };
  float*  net0    = (float*)carve((size_t)NN * 256 * 4);
  float*  netbuf  = (float*)carve((size_t)NN * HID * 4);
  float*  featbuf = (float*)carve((size_t)NN * FSTRIDE * 4);
  float*  accbuf  = (float*)carve((size_t)NN * HID * 4);
  __bf16* w0p[5]; __bf16* wsp[5]; __bf16* w1p[5];
  const int KBb[5] = {16, 17, 17, 17, 17};  // ceil(2F/32): 512, 544x4
  const int Fb[5]  = {256, 259, 259, 259, 259};
  for (int i = 0; i < 5; ++i) {
    w0p[i] = (__bf16*)carve((size_t)KBb[i] * 4096 * 2);
    wsp[i] = (__bf16*)carve((size_t)KBb[i] * 4096 * 2);
    w1p[i] = (__bf16*)carve((size_t)4 * 4096 * 2);
  }
  __bf16* wcp = (__bf16*)carve((size_t)4 * 4096 * 2);

  // ---- pack weights into fragment order ----
  for (int i = 0; i < 5; ++i) {
    pack_weights_kernel<<<KBb[i] * 16, 256, 0, stream>>>(w0[i], w0p[i], 2 * Fb[i], KBb[i]);
    pack_weights_kernel<<<KBb[i] * 16, 256, 0, stream>>>(wsc[i], wsp[i], 2 * Fb[i], KBb[i]);
    pack_weights_kernel<<<4 * 16, 256, 0, stream>>>(w1[i], w1p[i], HID, 4);
  }
  pack_weights_kernel<<<4 * 16, 256, 0, stream>>>(fcw, wcp, HID, 4);

  // ---- input embedding ----
  fcpos_kernel<<<NN, 256, 0, stream>>>(p, fposw, fposb, net0);

  // ---- block 0: gather source = net0 [NN,256] ----
  init_neg_inf_kernel<<<(NN * HID) / 256, 256, 0, stream>>>(accbuf, NN * HID);
  edgeconv_kernel<<<NE / TM, THREADS, 0, stream>>>(
      net0, 256, 256, src, dst, w0p[0], wsp[0], w1p[0], b0[0], b1[0], KBb[0], accbuf);
  fixup_kernel<<<NN, 128, 0, stream>>>(accbuf, p, netbuf, featbuf);

  // ---- blocks 1..4: gather source = featbuf [NN,260], F=259 ----
  for (int i = 1; i < 5; ++i) {
    init_neg_inf_kernel<<<(NN * HID) / 256, 256, 0, stream>>>(accbuf, NN * HID);
    edgeconv_kernel<<<NE / TM, THREADS, 0, stream>>>(
        featbuf, FSTRIDE, 259, src, dst, w0p[i], wsp[i], w1p[i], b0[i], b1[i],
        KBb[i], accbuf);
    fixup_kernel<<<NN, 128, 0, stream>>>(accbuf, p, netbuf, featbuf);
  }

  // ---- output head ----
  final_wmma_kernel<<<NN / TM, THREADS, 0, stream>>>(netbuf, wcp, fcb, (float*)d_out);
}
